// Attention_32220844654630
// MI455X (gfx1250) — matrix operand order repaired and checked by reading, not compiled, not run on hardware
//
#include <hip/hip_runtime.h>

// ---------------- problem constants (match reference) ----------------
constexpr int H   = 128;
constexpr int NP  = 1000;
constexpr int NV  = 50000;
constexpr int E   = 600000;

// ---------------- types ----------------
typedef __attribute__((ext_vector_type(16))) __bf16 v16bf;
typedef __attribute__((ext_vector_type(8)))  float  v8f;
typedef int v4i_vs __attribute__((vector_size(4 * sizeof(int))));  // matches builtin param type

#if __has_builtin(__builtin_amdgcn_global_load_async_to_lds_b128)
#define HAVE_ASYNC_LDS 1
#else
#define HAVE_ASYNC_LDS 0
#endif

#if __has_builtin(__builtin_amdgcn_cvt_pk_bf16_f32)
#define HAVE_CVT_PK_BF16 1
#else
#define HAVE_CVT_PK_BF16 0
#endif

// pack two floats into one dword of bf16 (RNE) — used only in prep kernels
__device__ __forceinline__ unsigned pack_bf2(float lo, float hi) {
#if HAVE_CVT_PK_BF16
    return __builtin_bit_cast(unsigned, __builtin_amdgcn_cvt_pk_bf16_f32(lo, hi));
#else
    unsigned ul = __float_as_uint(lo), uh = __float_as_uint(hi);
    ul = (ul + 0x7FFFu + ((ul >> 16) & 1u)) >> 16;
    uh = (uh + 0x7FFFu + ((uh >> 16) & 1u)) & 0xFFFF0000u;
    return uh | ul;
#endif
}

__device__ __forceinline__ __bf16 f2bf(float f) {
    unsigned u = __float_as_uint(f);
    unsigned short hh = (unsigned short)((u + 0x7FFFu + ((u >> 16) & 1u)) >> 16);
    return __builtin_bit_cast(__bf16, hh);
}

__device__ __forceinline__ float fast_tanh(float x) {
#if __has_builtin(__builtin_amdgcn_tanhf)
    return __builtin_amdgcn_tanhf(x);
#elif __has_builtin(__builtin_amdgcn_tanh_f32)
    return __builtin_amdgcn_tanh_f32(x);
#else
    return tanhf(x);
#endif
}

__device__ __forceinline__ float wave_sum(float x) {
    #pragma unroll
    for (int off = 16; off > 0; off >>= 1) x += __shfl_xor(x, off, 32);
    return x;
}
__device__ __forceinline__ float wave_max(float x) {
    #pragma unroll
    for (int off = 16; off > 0; off >>= 1) x = fmaxf(x, __shfl_xor(x, off, 32));
    return x;
}

// ---------------- weight prep: f32 row-major -> bf16 col-major ----------------
__global__ void prep_weights(const float* __restrict__ Wv, const float* __restrict__ Wp,
                             const float* __restrict__ W1,
                             __bf16* __restrict__ WvT, __bf16* __restrict__ WpT,
                             __bf16* __restrict__ W1T) {
    int i = blockIdx.x * blockDim.x + threadIdx.x;
    if (i < H * H) {                       // WvT[n][k] = Wv[k][n]
        int n = i & 127, k = i >> 7;
        WvT[n * H + k] = f2bf(Wv[k * H + n]);
    } else if (i < 2 * H * H) {            // WpT
        int j = i - H * H; int n = j & 127, k = j >> 7;
        WpT[n * H + k] = f2bf(Wp[k * H + n]);
    } else if (i < 2 * H * H + H * 64) {   // W1T[n][k] = W1[k][n], n<64
        int j = i - 2 * H * H; int n = j % 64, k = j / 64;
        W1T[n * H + k] = f2bf(W1[k * 64 + n]);
    }
}

// ---------------- activation prep: V,P f32 -> bf16 panels (one pass, paid once) ----------------
__global__ void prep_feats(const float* __restrict__ V, const float* __restrict__ P,
                           unsigned* __restrict__ Vbf, unsigned* __restrict__ Pbf) {
    int i = blockIdx.x * blockDim.x + threadIdx.x;
    const int NVP = NV * H / 2;            // 3,200,000 dword pairs
    const int NPP = NP * H / 2;            // 64,000
    if (i < NVP) {
        float2 x = ((const float2*)V)[i];
        Vbf[i] = pack_bf2(x.x, x.y);
    } else if (i < NVP + NPP) {
        int j = i - NVP;
        float2 x = ((const float2*)P)[j];
        Pbf[j] = pack_bf2(x.x, x.y);
    }
}

// ---------------- one 16x16 output tile, K=128, bf16 WMMA ----------------
// A: bf16 row-major [M x 128]; WT: bf16 [cols x 128] column-major (may live in LDS)
__device__ __forceinline__ v8f wmma_tile_k128(const __bf16* __restrict__ A, int rowBase, int M,
                                              const __bf16* __restrict__ WT, int colBase) {
    const int lane = threadIdx.x & 31;
    const int h = lane >> 4;                // half-wave
    const int n = lane & 15;                // output column within tile
    int row = rowBase + (lane & 15);        // A row per lane
    if (row >= M) row = M - 1;              // clamp (stores are guarded by caller)
    const __bf16* __restrict__ Ar = A + (size_t)row * H;
    const __bf16* __restrict__ Bc = WT + (size_t)(colBase + n) * H;
    v8f c = {};
    #pragma unroll
    for (int kb = 0; kb < H; kb += 32) {
        // A lane layout: element e -> K = kb + h*16 + e
        union { uint4 q[2]; v16bf v; } a;
        a.q[0] = *(const uint4*)(Ar + kb + h * 16);
        a.q[1] = *(const uint4*)(Ar + kb + h * 16 + 8);
        // B lane layout: element e -> K = kb + h*16 + e (contiguous 16 bf16 = 32B)
        v16bf b = *(const v16bf*)(Bc + kb + h * 16);
        c = __builtin_amdgcn_wmma_f32_16x16x32_bf16(false, a.v, false, b, (short)0, c, false, false);
    }
    return c;
}

// unguarded epilogue: base address + 8 immediate-offset stores
__device__ __forceinline__ void store_tile_nog(float* __restrict__ out, int ldo, int rowBase,
                                               const float* __restrict__ bias, v8f c) {
    const int lane = threadIdx.x & 31;
    const int h = lane >> 4, n = lane & 15;
    float* p = out + (size_t)(rowBase + 8 * h) * ldo + n;
    const float bn = bias[n];
    #pragma unroll
    for (int r = 0; r < 8; ++r) p[(size_t)r * ldo] = c[r] + bn;
}

__device__ __forceinline__ void store_tile_guard(float* __restrict__ out, int ldo, int rowBase, int M,
                                                 const float* __restrict__ bias, v8f c) {
    const int lane = threadIdx.x & 31;
    const int h = lane >> 4, n = lane & 15;
    const float bn = bias[n];
    #pragma unroll
    for (int r = 0; r < 8; ++r) {
        int m = rowBase + r + 8 * h;
        if (m < M) out[(size_t)m * ldo + n] = c[r] + bn;
    }
}

// ---------------- fused voxel GEMM: Xv = V@Wv+bv (8 tiles), T = V@W1+b1 (4 tiles) ------------
// One block per 16-row M-tile; 12 waves = 12 N-tiles. The 48KB bf16 weight panel
// (WvT || W1T, contiguous) is staged into LDS with async global->LDS loads.
__global__ __launch_bounds__(384) void gemm_voxel(const __bf16* __restrict__ Vbf,
                                                  const __bf16* __restrict__ Wcat, // 49152 B
                                                  const float* __restrict__ bv,
                                                  const float* __restrict__ b1,
                                                  float* __restrict__ Xv,
                                                  float* __restrict__ T) {
    __shared__ __align__(32) char smem[49152];
    const int t = threadIdx.y * 32 + threadIdx.x;   // 0..383, 128B per thread
#if HAVE_ASYNC_LDS
    {
        char* src = (char*)Wcat;
        char* dst = (char*)smem;
        #pragma unroll
        for (int i = 0; i < 8; ++i) {
            __builtin_amdgcn_global_load_async_to_lds_b128(
                (v4i_vs*)(src + (size_t)t * 128 + i * 16),
                (v4i_vs*)(dst + t * 128 + i * 16),
                0, 0);
        }
#if __has_builtin(__builtin_amdgcn_s_wait_asynccnt)
        __builtin_amdgcn_s_wait_asynccnt(0);
#else
        asm volatile("s_wait_asynccnt 0" ::: "memory");
#endif
    }
#else
    {
        const float4* s4 = (const float4*)Wcat;
        float4* d4 = (float4*)smem;
        #pragma unroll
        for (int i = 0; i < 8; ++i) d4[t + 384 * i] = s4[t + 384 * i];
    }
#endif
    __syncthreads();

    const int mtile = blockIdx.x;
    const int ntile = threadIdx.y;                  // wave-uniform 0..11
    const int rowBase = mtile * 16;
    const __bf16* B = (const __bf16*)smem;
    if (ntile < 8) {
        v8f c = wmma_tile_k128(Vbf, rowBase, NV, B, ntile * 16);
        store_tile_nog(Xv + ntile * 16, H, rowBase, bv + ntile * 16, c);
    } else {
        int nt = ntile - 8;
        v8f c = wmma_tile_k128(Vbf, rowBase, NV, B + 16384 /* 32768B */, nt * 16);
        store_tile_nog(T + nt * 16, 64, rowBase, b1 + nt * 16, c);
    }
}

// ---------------- program GEMM: Xp = P@Wp+bp ----------------
__global__ __launch_bounds__(256) void gemm_program(const __bf16* __restrict__ Pbf,
                                                    const __bf16* __restrict__ WpT,
                                                    const float* __restrict__ bp,
                                                    float* __restrict__ Xp) {
    const int MT = (NP + 15) / 16;          // 63
    int w = blockIdx.x * 8 + threadIdx.y;
    if (w >= MT * 8) return;
    int mtile = w / 8, ntile = w % 8;
    v8f c = wmma_tile_k128(Pbf, mtile * 16, NP, WpT, ntile * 16);
    store_tile_guard(Xp + ntile * 16, H, mtile * 16, NP, bp + ntile * 16, c);
}

// ---------------- mask head: logits = T@W2+b2, gumbel softmax (tau=1) ----------------
__global__ void mask_kernel(const float* __restrict__ T, const float* __restrict__ W2,
                            const float* __restrict__ b2, const float* __restrict__ g_mask,
                            float* __restrict__ out_hard, float* __restrict__ out_soft,
                            float* __restrict__ sm0) {
    int v = blockIdx.x * blockDim.x + threadIdx.x;
    if (v >= NV) return;
    float l0 = b2[0], l1 = b2[1];
    const float* t = T + (size_t)v * 64;
    #pragma unroll 8
    for (int k = 0; k < 64; ++k) {
        float tv = t[k];
        l0 = fmaf(tv, W2[2 * k], l0);
        l1 = fmaf(tv, W2[2 * k + 1], l1);
    }
    float y0 = l0 + g_mask[2 * v], y1 = l1 + g_mask[2 * v + 1];
    float m = fmaxf(y0, y1);
    float e0 = __expf(y0 - m), e1 = __expf(y1 - m);
    float s0 = e0 / (e0 + e1);
    out_hard[v] = (y0 >= y1) ? 1.0f : 0.0f;   // one_hot(argmax)[:,0], first-index ties
    out_soft[v] = s0;
    sm0[v]      = s0;
}

// ---------------- per-edge attention score: y[e] = theta . tanh(Xv[vi]+Xp[pj]) + g ----------------
__global__ __launch_bounds__(256) void edge_att(const float* __restrict__ Xv,
                                                const float* __restrict__ Xp,
                                                const int* __restrict__ pidx,
                                                const int* __restrict__ vidx,
                                                const float* __restrict__ theta,
                                                const float* __restrict__ g_att,
                                                float* __restrict__ y) {
    int e = blockIdx.x * 8 + threadIdx.y;    // one wave per edge
    if (e >= E) return;
    int lane = threadIdx.x;
    int vi = vidx[e], pj = pidx[e];
    float4 xv = ((const float4*)(Xv + (size_t)vi * H))[lane];
    float4 xp = ((const float4*)(Xp + (size_t)pj * H))[lane];
    float4 th = ((const float4*)theta)[lane];
    float s = th.x * fast_tanh(xv.x + xp.x) + th.y * fast_tanh(xv.y + xp.y)
            + th.z * fast_tanh(xv.z + xp.z) + th.w * fast_tanh(xv.w + xp.w);
    s = wave_sum(s);
    if (lane == 0) y[e] = s + g_att[e];
}

// ---------------- CSR starts from sorted voxel index ----------------
__global__ void build_starts(const int* __restrict__ vidx, int* __restrict__ start) {
    int e = blockIdx.x * blockDim.x + threadIdx.x;
    if (e > E) return;
    int v  = (e < E) ? vidx[e] : NV;
    int pv = (e > 0) ? vidx[e - 1] : -1;
    for (int u = pv + 1; u <= v; ++u) start[u] = e;
}

// ---------------- per-voxel segment softmax + weighted scatter ----------------
__global__ __launch_bounds__(256) void per_voxel(const float* __restrict__ y,
                                                 const int* __restrict__ start,
                                                 const int* __restrict__ pidx,
                                                 const float* __restrict__ P,
                                                 const float* __restrict__ V,
                                                 const float* __restrict__ sm0,
                                                 float* __restrict__ out_hard_att,
                                                 float* __restrict__ out_soft_att,
                                                 float* __restrict__ out_new) {
    int v = blockIdx.x * 8 + threadIdx.y;    // one wave per voxel
    if (v >= NV) return;
    int lane = threadIdx.x;
    int n0 = start[v], n1 = start[v + 1];

    float mx = -3.4028235e38f;
    for (int e = n0 + lane; e < n1; e += 32) mx = fmaxf(mx, y[e]);
    mx = wave_max(mx);

    float s = 0.0f;
    for (int e = n0 + lane; e < n1; e += 32) s += __expf(y[e] - mx);
    s = wave_sum(s);
    float inv_s = (n1 > n0) ? 1.0f / s : 0.0f;

    for (int e = n0 + lane; e < n1; e += 32) {
        float ye = y[e];
        out_soft_att[e] = __expf(ye - mx) * inv_s;
        out_hard_att[e] = (ye == mx) ? 1.0f : 0.0f;   // straight-through fwd == indicator
    }

    // agg[h] = sum_e soft_att[e] * P[pidx[e], h]; lane owns 4 channels
    float4 acc = make_float4(0.f, 0.f, 0.f, 0.f);
    for (int e = n0; e < n1; ++e) {
        float sa = __expf(y[e] - mx) * inv_s;         // wave-uniform
        float4 pv4 = ((const float4*)(P + (size_t)pidx[e] * H))[lane];
        acc.x = fmaf(sa, pv4.x, acc.x);
        acc.y = fmaf(sa, pv4.y, acc.y);
        acc.z = fmaf(sa, pv4.z, acc.z);
        acc.w = fmaf(sa, pv4.w, acc.w);
    }
    float sm = sm0[v];
    float4 vf = ((const float4*)(V + (size_t)v * H))[lane];
    float4 o;
    o.x = fmaf(sm, acc.x, vf.x);
    o.y = fmaf(sm, acc.y, vf.y);
    o.z = fmaf(sm, acc.z, vf.z);
    o.w = fmaf(sm, acc.w, vf.w);
    ((float4*)(out_new))[(size_t)v * (H / 4) + lane] = o;
}

// ---------------- host launcher ----------------
extern "C" void kernel_launch(void* const* d_in, const int* in_sizes, int n_in,
                              void* d_out, int out_size, void* d_ws, size_t ws_size,
                              hipStream_t stream) {
    const float* P      = (const float*)d_in[0];   // [NP,H]
    const float* V      = (const float*)d_in[1];   // [NV,H]
    const int*   pidx   = (const int*)  d_in[2];   // [E]
    const int*   vidx   = (const int*)  d_in[3];   // [E] sorted
    const float* W1     = (const float*)d_in[4];   // [H,64]
    const float* b1     = (const float*)d_in[5];
    const float* W2     = (const float*)d_in[6];   // [64,2]
    const float* b2     = (const float*)d_in[7];
    const float* Wv     = (const float*)d_in[8];   // [H,H]
    const float* bv     = (const float*)d_in[9];
    const float* Wp     = (const float*)d_in[10];  // [H,H]
    const float* bp     = (const float*)d_in[11];
    const float* theta  = (const float*)d_in[12];  // [H]
    const float* g_mask = (const float*)d_in[13];  // [NV,2]
    const float* g_att  = (const float*)d_in[14];  // [E]

    float* out = (float*)d_out;
    float* out_hard     = out;                 // [NV]
    float* out_soft     = out + NV;            // [NV]
    float* out_hard_att = out + 2 * NV;        // [E]
    float* out_soft_att = out + 2 * NV + E;    // [E]
    float* out_new      = out + 2 * NV + 2 * E;// [NV,H]

    // workspace carve (offsets 256B-aligned; WvT and W1T intentionally contiguous)
    char* ws = (char*)d_ws;
    size_t o = 0;
    auto carve = [&](size_t bytes) { void* p = ws + o; o += (bytes + 255) & ~(size_t)255; return p; };
    __bf16* WvT  = (__bf16*)carve(H * H * sizeof(__bf16));    // 32768 B
    __bf16* W1T  = (__bf16*)carve(H * 64 * sizeof(__bf16));   // 16384 B (== WvT + 16384 elems)
    __bf16* WpT  = (__bf16*)carve(H * H * sizeof(__bf16));
    unsigned* Vbf = (unsigned*)carve((size_t)NV * H / 2 * sizeof(unsigned)); // bf16 panel
    unsigned* Pbf = (unsigned*)carve((size_t)NP * H / 2 * sizeof(unsigned)); // bf16 panel
    float*  Xv   = (float*) carve((size_t)NV * H * sizeof(float));
    float*  T    = (float*) carve((size_t)NV * 64 * sizeof(float));
    float*  Xp   = (float*) carve((size_t)NP * H * sizeof(float));
    float*  ybuf = (float*) carve((size_t)E * sizeof(float));
    float*  sm0  = (float*) carve((size_t)NV * sizeof(float));
    int*    strt = (int*)   carve((size_t)(NV + 1) * sizeof(int));

    // 1. weights -> bf16 column-major ; activations -> bf16 row-major
    prep_weights<<<(2 * H * H + H * 64 + 255) / 256, 256, 0, stream>>>(Wv, Wp, W1, WvT, WpT, W1T);
    prep_feats<<<((NV + NP) * H / 2 + 255) / 256, 256, 0, stream>>>(V, P, Vbf, Pbf);

    // 2. WMMA GEMMs (voxel GEMM: 1 block per M-tile, 12 waves, async-LDS weight panel)
    gemm_voxel<<<NV / 16, dim3(32, 12), 0, stream>>>((const __bf16*)Vbf, WvT, bv, b1, Xv, T);
    {
        int waves = ((NP + 15) / 16) * 8;
        gemm_program<<<(waves + 7) / 8, dim3(32, 8), 0, stream>>>((const __bf16*)Pbf, WpT, bp, Xp);
    }

    // 3. mask head
    mask_kernel<<<(NV + 255) / 256, 256, 0, stream>>>(T, W2, b2, g_mask, out_hard, out_soft, sm0);

    // 4. per-edge attention scores
    edge_att<<<(E + 7) / 8, dim3(32, 8), 0, stream>>>(Xv, Xp, pidx, vidx, theta, g_att, ybuf);

    // 5. CSR starts from sorted voxel index
    build_starts<<<(E + 1 + 255) / 256, 256, 0, stream>>>(vidx, strt);

    // 6. segment softmax + weighted scatter + residual
    per_voxel<<<(NV + 7) / 8, dim3(32, 8), 0, stream>>>(ybuf, strt, pidx, P, V, sm0,
                                                        out_hard_att, out_soft_att, out_new);
}
